// UMT5Attention_3607772528880
// MI455X (gfx1250) — compile-verified, matrix operand order checked by reading, not run on hardware
//
#include <hip/hip_runtime.h>
#include <hip/hip_bf16.h>

// ---------------------------------------------------------------------------
// UMT5 attention for MI455X (gfx1250, wave32, WMMA, async global->LDS).
// Pipeline:
//   1) cvt hidden fp32 -> bf16            (8 MB, one-time)
//   2) transpose+cvt Wq/Wk/Wv/Wo -> bf16 Wt[n][k] (2 MB each, one-time)
//   3) 3x WMMA GEMM (async double-buffered tiles) -> Q,K,V bf16
//   4) flash attention (WMMA QK^T + online softmax + WMMA PV)
//   5) WMMA GEMM O x Wo^T -> fp32 out
// Workspace (bf16): hidden | Wq^T Wk^T Wv^T Wo^T | Q K V O  = 48 MB.
// ---------------------------------------------------------------------------

typedef __attribute__((ext_vector_type(16))) __bf16 v16bf;
typedef __attribute__((ext_vector_type(8)))  float  v8f;

#define S_LEN   2048
#define D_MODEL 1024
#define NHEADS  16
#define CKV     64

// fp32 -> bf16 round-to-nearest-even
static __device__ __forceinline__ unsigned short f2bf(float f) {
    union { float f; unsigned u; } x; x.f = f;
    unsigned r = x.u + 0x7FFFu + ((x.u >> 16) & 1u);
    return (unsigned short)(r >> 16);
}

// 16-byte async copy global -> LDS (GLOBAL_LOAD_ASYNC_TO_LDS_B128, ASYNCcnt).
// vdst = 32-bit LDS byte address (low 32 bits of generic shared pointer),
// vaddr = 64-bit global address (GV mode, saddr = off).
static __device__ __forceinline__ void async_b128(void* lds, const void* gsrc) {
    unsigned loff = (unsigned)(unsigned long long)lds;
    asm volatile("global_load_async_to_lds_b128 %0, %1, off"
                 :: "v"(loff), "v"(gsrc) : "memory");
}

// Load a 16x32 bf16 A-fragment (or B-fragment via a K-contiguous transposed
// tile) per ISA 7.12.2: lanes 0-15 K={0..7,16..23}, lanes 16-31 K={8..15,24..31}.
static __device__ __forceinline__ v16bf ld_frag(const unsigned short* rowPtr, int hb) {
    union { unsigned int u[8]; v16bf v; } f;
    const unsigned int* p = (const unsigned int*)(rowPtr + hb);
#pragma unroll
    for (int v = 0; v < 4; ++v) f.u[v] = p[v];         // K = hb + {0,2,4,6}
#pragma unroll
    for (int v = 0; v < 4; ++v) f.u[4 + v] = p[8 + v]; // K = hb + 16 + {0,2,4,6}
    return f.v;
}

static __device__ __forceinline__ v8f wmma_bf16(v16bf a, v16bf b, v8f c) {
    return __builtin_amdgcn_wmma_f32_16x16x32_bf16(
        false, a, false, b, (short)0, c, false, false);
}

// T5 relative position bucket (matches the jnp reference).
static __device__ __forceinline__ int rel_bucket(int rel) {
    const int   nb        = 16;                  // NUM_BUCKETS / 2
    const int   max_exact = 8;
    const float inv_log   = 0.3606737602222409f; // 1 / ln(16)
    int rb = (rel > 0) ? nb : 0;
    int ap = rel < 0 ? -rel : rel;
    int am = ap > 1 ? ap : 1;
    int large = max_exact +
        (int)(__logf((float)am * 0.125f) * inv_log * (float)(nb - max_exact));
    if (large > nb - 1) large = nb - 1;
    return rb + (ap < max_exact ? ap : large);
}

// ---------------------------------------------------------------------------
// One-time prep kernels (memory-bound, ~1.6us total at 23.3 TB/s)
// ---------------------------------------------------------------------------
__global__ __launch_bounds__(256) void cvt_f32_bf16(const float* __restrict__ src,
                                                    unsigned short* __restrict__ dst) {
    int i = blockIdx.x * 256 + threadIdx.x;          // one float4 per thread
    float4 f = ((const float4*)src)[i];
    ushort4 o;
    o.x = f2bf(f.x); o.y = f2bf(f.y); o.z = f2bf(f.z); o.w = f2bf(f.w);
    ((ushort4*)dst)[i] = o;
}

// W (K x N fp32, K=N=1024) -> Wt (N x K bf16)
__global__ __launch_bounds__(256) void transpose_cvt_bf16(const float* __restrict__ W,
                                                          unsigned short* __restrict__ Wt) {
    __shared__ float tile[32][33];
    const int k0 = blockIdx.y * 32, n0 = blockIdx.x * 32;
    const int tx = threadIdx.x, ty = threadIdx.y;    // block (32, 8)
#pragma unroll
    for (int i = ty; i < 32; i += 8)
        tile[i][tx] = W[(size_t)(k0 + i) * D_MODEL + (n0 + tx)];   // tile[k][n]
    __syncthreads();
#pragma unroll
    for (int i = ty; i < 32; i += 8)
        Wt[(size_t)(n0 + i) * D_MODEL + (k0 + tx)] = f2bf(tile[tx][i]);
}

// ---------------------------------------------------------------------------
// WMMA GEMM: C(MxN) = A(MxK bf16, row-major) * Bt(NxK bf16)^T
// 256 threads / 8 waves, C tile 128x128, wave subtile 32x64,
// double-buffered async global->LDS K-tiles (step 32).
// ---------------------------------------------------------------------------
template <bool OUT_BF16>
__global__ __launch_bounds__(256) void gemm_bf16_async(const unsigned short* __restrict__ A,
                                                       const unsigned short* __restrict__ Bt,
                                                       void* __restrict__ Cp,
                                                       int M, int N, int K) {
    // stride 40 ushorts = 80B: 16B-aligned async dests + bank-offset padding
    __shared__ __align__(16) unsigned short As[2][128][40];
    __shared__ __align__(16) unsigned short Bs[2][128][40];

    const int tid  = threadIdx.x;
    const int lane = tid & 31, wave = tid >> 5;
    const int hb   = (lane >> 4) * 8, lm = lane & 15;
    const int m0 = blockIdx.y * 128, n0 = blockIdx.x * 128;
    const int mBase = (wave >> 1) * 32;   // 4 waves along M
    const int nBase = (wave & 1) * 64;    // 2 waves along N

    v8f acc[2][4];
    v8f zero = {};
#pragma unroll
    for (int mb = 0; mb < 2; ++mb)
#pragma unroll
        for (int nb = 0; nb < 4; ++nb) acc[mb][nb] = zero;

    // Each tile: 128 rows x 64B = 512 x b128 -> 2 chunks/thread per matrix.
    auto issue_tile = [&](int k0, int buf) {
#pragma unroll
        for (int t = 0; t < 2; ++t) {
            int u = tid + t * 256;
            int row = u >> 2, part = u & 3;                     // 4 x 16B per row
            async_b128(&As[buf][row][part * 8],
                       A + (size_t)(m0 + row) * K + k0 + part * 8);
            async_b128(&Bs[buf][row][part * 8],
                       Bt + (size_t)(n0 + row) * K + k0 + part * 8);
        }
    };

    issue_tile(0, 0);
    const int nk = K / 32;
    for (int t = 0; t < nk; ++t) {
        const int buf = t & 1;
        __syncthreads();                       // everyone done reading buf^1
        if (t + 1 < nk) {
            issue_tile((t + 1) * 32, buf ^ 1); // prefetch next tile
            asm volatile("s_wait_asynccnt 0x4" ::: "memory"); // current tile landed
        } else {
            asm volatile("s_wait_asynccnt 0x0" ::: "memory");
        }
        __syncthreads();                       // all waves' copies of buf landed

        v16bf afr[2], bfr[4];
#pragma unroll
        for (int mb = 0; mb < 2; ++mb)
            afr[mb] = ld_frag(&As[buf][mBase + mb * 16 + lm][0], hb);
#pragma unroll
        for (int nb = 0; nb < 4; ++nb)
            bfr[nb] = ld_frag(&Bs[buf][nBase + nb * 16 + lm][0], hb);
#pragma unroll
        for (int mb = 0; mb < 2; ++mb)
#pragma unroll
            for (int nb = 0; nb < 4; ++nb)
                acc[mb][nb] = wmma_bf16(afr[mb], bfr[nb], acc[mb][nb]);
    }

    // C/D layout: VGPR r -> M = r + 8*(lane>=16); N = lane%16
#pragma unroll
    for (int mb = 0; mb < 2; ++mb)
#pragma unroll
        for (int nb = 0; nb < 4; ++nb)
#pragma unroll
            for (int r = 0; r < 8; ++r) {
                int row = m0 + mBase + mb * 16 + r + hb;
                int col = n0 + nBase + nb * 16 + lm;
                float vv = acc[mb][nb][r];
                if (OUT_BF16) ((unsigned short*)Cp)[(size_t)row * N + col] = f2bf(vv);
                else          ((float*)Cp)[(size_t)row * N + col] = vv;
            }
}

// ---------------------------------------------------------------------------
// Flash attention: grid (S/64, NHEADS, B), block 128 (4 waves).
// Each wave owns a 16-row Q strip; 32 keys per step through LDS.
// K tile arrives via async b128 copies; V is transposed through VGPRs.
// ---------------------------------------------------------------------------
__global__ __launch_bounds__(128) void attn_kernel(const unsigned short* __restrict__ q,
                                                   const unsigned short* __restrict__ k,
                                                   const unsigned short* __restrict__ v,
                                                   const float* __restrict__ rel_bias,
                                                   const int* __restrict__ mask,
                                                   unsigned short* __restrict__ o) {
    __shared__ float biasLUT[2 * S_LEN];                       // 16KB
    __shared__ __align__(16) unsigned short Kt[32][CKV + 8];   // [j][c], 144B rows
    __shared__ unsigned short VtT[CKV][34];                    // [c][j]
    __shared__ unsigned short Pst[4][16][34];                  // per-wave P staging

    const int tid  = threadIdx.x;
    const int lane = tid & 31, wave = tid >> 5;
    const int hb   = (lane >> 4) * 8, lm = lane & 15;
    const int head = blockIdx.y, b = blockIdx.z;
    const int qBase = blockIdx.x * 64 + wave * 16;

    // bias LUT once per block: log-bucket out of the hot loop
    for (int idx = tid; idx < 2 * S_LEN - 1; idx += 128) {
        int rel = idx - (S_LEN - 1);                           // rel = j - i
        biasLUT[idx] = rel_bias[rel_bucket(rel) * NHEADS + head];
    }

    // Q strip -> two 16x32 A-fragments kept in registers
    v16bf aq[2];
    {
        const unsigned short* qrow =
            q + ((size_t)(b * S_LEN + qBase + lm)) * D_MODEL + head * CKV;
        aq[0] = ld_frag(qrow, hb);
        aq[1] = ld_frag(qrow + 32, hb);
    }

    v8f oacc[4];
    v8f zero = {};
#pragma unroll
    for (int nb = 0; nb < 4; ++nb) oacc[nb] = zero;
    float mrow[8], lrow[8];
#pragma unroll
    for (int r = 0; r < 8; ++r) { mrow[r] = -3.0e38f; lrow[r] = 0.0f; }

    __syncthreads();  // biasLUT ready

    for (int j0 = 0; j0 < S_LEN; j0 += 32) {
        __syncthreads();  // prev iteration done reading Kt/VtT

        // K tile: 32 rows x 128B = 256 b128 chunks / 128 threads
#pragma unroll
        for (int t = 0; t < 2; ++t) {
            int u = tid + t * 128;
            int j = u >> 3, part = u & 7;
            async_b128(&Kt[j][part * 8],
                       k + ((size_t)(b * S_LEN + j0 + j)) * D_MODEL
                         + head * CKV + part * 8);
        }
        // V tile transposed: [c][j]
#pragma unroll
        for (int it = 0; it < 8; ++it) {
            int u = tid + it * 128;
            int j = u >> 5, cp = (u & 31) * 2;
            unsigned int vw = *(const unsigned int*)(
                v + ((size_t)(b * S_LEN + j0 + j)) * D_MODEL + head * CKV + cp);
            VtT[cp][j]     = (unsigned short)(vw & 0xFFFFu);
            VtT[cp + 1][j] = (unsigned short)(vw >> 16);
        }
        asm volatile("s_wait_asynccnt 0x0" ::: "memory");
        __syncthreads();

        // Scores: S[16 x 32] = Q(16x64) . K^T
        v8f sc[2];
#pragma unroll
        for (int t = 0; t < 2; ++t) {
            const unsigned short* krow = &Kt[t * 16 + lm][0];
            v8f z = zero;
            z = wmma_bf16(aq[0], ld_frag(krow, hb), z);
            z = wmma_bf16(aq[1], ld_frag(krow + 32, hb), z);
            sc[t] = z;
        }

        // bias + mask (reference: bias -> -FLT_MAX where mask==0)
#pragma unroll
        for (int t = 0; t < 2; ++t) {
            int j = j0 + t * 16 + lm;
            bool ok = mask[b * S_LEN + j] != 0;
#pragma unroll
            for (int r = 0; r < 8; ++r) {
                int i = qBase + r + hb;
                sc[t][r] += ok ? biasLUT[j - i + (S_LEN - 1)] : -3.4028235e38f;
            }
        }

        // Online softmax; rows live in 16-lane halves -> width-16 shuffles
#pragma unroll
        for (int r = 0; r < 8; ++r) {
            float tm = fmaxf(sc[0][r], sc[1][r]);
#pragma unroll
            for (int off = 1; off < 16; off <<= 1)
                tm = fmaxf(tm, __shfl_xor(tm, off, 16));
            float nm    = fmaxf(mrow[r], tm);
            float scale = __expf(mrow[r] - nm);
            float p0 = __expf(sc[0][r] - nm);
            float p1 = __expf(sc[1][r] - nm);
            float ps = p0 + p1;
#pragma unroll
            for (int off = 1; off < 16; off <<= 1)
                ps += __shfl_xor(ps, off, 16);
            lrow[r] = lrow[r] * scale + ps;
            mrow[r] = nm;
#pragma unroll
            for (int nb = 0; nb < 4; ++nb) oacc[nb][r] *= scale;
            int prow = r + hb;
            Pst[wave][prow][lm]      = f2bf(p0);
            Pst[wave][prow][16 + lm] = f2bf(p1);
        }
        asm volatile("s_wait_dscnt 0x0" ::: "memory");  // P visible intra-wave

        // O += P(16x32) . V(32x64)
        v16bf ap = ld_frag(&Pst[wave][lm][0], hb);
#pragma unroll
        for (int nb = 0; nb < 4; ++nb)
            oacc[nb] = wmma_bf16(ap, ld_frag(&VtT[nb * 16 + lm][0], hb), oacc[nb]);
    }

    // Normalize; write O bf16 in [b][s][head*64+c] for the Wo GEMM
#pragma unroll
    for (int nb = 0; nb < 4; ++nb)
#pragma unroll
        for (int r = 0; r < 8; ++r) {
            int i   = qBase + r + hb;
            int col = head * CKV + nb * 16 + lm;
            o[(size_t)(b * S_LEN + i) * D_MODEL + col] = f2bf(oacc[nb][r] / lrow[r]);
        }
}

// ---------------------------------------------------------------------------
extern "C" void kernel_launch(void* const* d_in, const int* in_sizes, int n_in,
                              void* d_out, int out_size, void* d_ws, size_t ws_size,
                              hipStream_t stream) {
    const float* hidden   = (const float*)d_in[0];
    const int*   mask     = (const int*)d_in[1];
    const float* Wq       = (const float*)d_in[2];
    const float* Wk       = (const float*)d_in[3];
    const float* Wv       = (const float*)d_in[4];
    const float* Wo       = (const float*)d_in[5];
    const float* rel_bias = (const float*)d_in[6];
    float*       out      = (float*)d_out;

    const int M = 2 * S_LEN;                    // 4096 flattened rows
    const int N = D_MODEL, K = D_MODEL;         // 1024
    const size_t me = (size_t)M * D_MODEL;      // 4M elements
    const size_t we = (size_t)D_MODEL * D_MODEL;

    unsigned short* hb  = (unsigned short*)d_ws;   // bf16 hidden     (8 MB)
    unsigned short* wqt = hb + me;                 // bf16 W^T x4     (2 MB each)
    unsigned short* wkt = wqt + we;
    unsigned short* wvt = wkt + we;
    unsigned short* wot = wvt + we;
    unsigned short* qb  = wot + we;                // bf16 Q,K,V,O    (8 MB each)
    unsigned short* kb  = qb + me;
    unsigned short* vb  = kb + me;
    unsigned short* ob  = vb + me;                 // ws total = 48 MB

    // one-time conversions
    cvt_f32_bf16<<<(int)(me / 4 / 256), 256, 0, stream>>>(hidden, hb);
    dim3 tb(32, 8), tg(D_MODEL / 32, D_MODEL / 32);
    transpose_cvt_bf16<<<tg, tb, 0, stream>>>(Wq, wqt);
    transpose_cvt_bf16<<<tg, tb, 0, stream>>>(Wk, wkt);
    transpose_cvt_bf16<<<tg, tb, 0, stream>>>(Wv, wvt);
    transpose_cvt_bf16<<<tg, tb, 0, stream>>>(Wo, wot);

    dim3 gg(N / 128, M / 128);                  // (8, 32)
    gemm_bf16_async<true><<<gg, 256, 0, stream>>>(hb, wqt, qb, M, N, K);
    gemm_bf16_async<true><<<gg, 256, 0, stream>>>(hb, wkt, kb, M, N, K);
    gemm_bf16_async<true><<<gg, 256, 0, stream>>>(hb, wvt, vb, M, N, K);

    attn_kernel<<<dim3(S_LEN / 64, NHEADS, 2), 128, 0, stream>>>(
        qb, kb, vb, rel_bias, mask, ob);

    gemm_bf16_async<false><<<gg, 256, 0, stream>>>(ob, wot, out, M, N, K);
}
